// FullAttention_8538394984504
// MI455X (gfx1250) — compile-verified, matrix operand order checked by reading, not run on hardware
//
#include <hip/hip_runtime.h>
#include <math.h>

// ---- CDNA5 flash-attention (causal): f16 WMMA, f32 online softmax.
// ---- 4 waves x 32 queries, double-buffered LDS pipeline, 1 barrier/iter.
// ---- Row-sum lane-merge deferred to epilogue (one shfl total).
typedef __attribute__((ext_vector_type(16))) _Float16 v16h;
typedef __attribute__((ext_vector_type(8)))  _Float16 v8h;
typedef __attribute__((ext_vector_type(4)))  _Float16 v4h;
typedef __attribute__((ext_vector_type(8)))  float    v8f;
typedef __attribute__((ext_vector_type(4)))  float    v4f;

constexpr int Bb = 2, Ls = 4096, Hh = 8, Dd = 64;
constexpr int BLOCK_M = 128;   // 4 waves x 32 query rows
constexpr int KB      = 32;    // keys per block (= WMMA K-dim for PV)
constexpr int KROW    = 72;    // LDS halves per K row  [32 x 64 + pad]
constexpr int VROW    = 40;    // LDS halves per V^T row [64 x 32 + pad]
constexpr float SCL   = 0.125f * 1.44269504088896340736f; // (1/sqrt(D))*log2e
constexpr float NEG   = -1.0e30f;

// raw v_exp_f32: no denormal-range fixup (masked scores -> exact 0)
__device__ inline float fexp2(float x) { return __builtin_amdgcn_exp2f(x); }

__device__ inline v16h frag(const _Float16* s, int o0, int o1) {
  v8h lo = *(const v8h*)(s + o0);     // 16B aligned by construction
  v8h hi = *(const v8h*)(s + o1);
  return __builtin_shufflevector(lo, hi, 0,1,2,3,4,5,6,7,8,9,10,11,12,13,14,15);
}

__global__ __launch_bounds__(128)
void fa_fwd_causal(const float* __restrict__ Q, const float* __restrict__ K,
                   const float* __restrict__ V, float* __restrict__ O) {
  __shared__ __align__(16) _Float16 sK[2][KB * KROW];  // [key][d]
  __shared__ __align__(16) _Float16 sV[2][Dd * VROW];  // [d][key] (transposed)

  const int tid  = threadIdx.x;
  const int wave = tid >> 5;
  const int lane = tid & 31;
  const int ll   = lane & 15;
  const int hb   = (lane >> 4) & 1;

  const int bh = blockIdx.y;
  const int b  = bh / Hh, h = bh % Hh;
  const int qbase = blockIdx.x * BLOCK_M;
  const int qt0 = qbase + wave * 32;      // this wave: queries qt0..qt0+31

  const size_t bstr = (size_t)Ls * Hh * Dd;
  const float* Qb = Q + (size_t)b * bstr + (size_t)h * Dd;
  const float* Kb = K + (size_t)b * bstr + (size_t)h * Dd;
  const float* Vb = V + (size_t)b * bstr + (size_t)h * Dd;
  float*       Ob = O + (size_t)b * bstr + (size_t)h * Dd;

  // cooperative-load mapping: float4 idx = i*128+tid -> (key, 4 d's), i=0..3
  int kk[4], dd[4];
#pragma unroll
  for (int i = 0; i < 4; ++i) {
    const int idx = i * 128 + tid;
    kk[i] = idx >> 4;
    dd[i] = (idx & 15) << 2;
  }

  // ---- Q^T B-fragments: [qi][dh], loaded once ----
  v16h bq[2][2];
#pragma unroll
  for (int qi = 0; qi < 2; ++qi) {
    const float* qrow = Qb + (size_t)(qt0 + qi * 16 + ll) * Hh * Dd;
#pragma unroll
    for (int f = 0; f < 2; ++f) {
      const int d0 = f * 32 + 8 * hb;
      v4f a0 = *(const v4f*)(qrow + d0);
      v4f a1 = *(const v4f*)(qrow + d0 + 4);
      v4f a2 = *(const v4f*)(qrow + d0 + 16);
      v4f a3 = *(const v4f*)(qrow + d0 + 20);
#pragma unroll
      for (int j = 0; j < 4; ++j) {
        bq[qi][f][j]      = (_Float16)a0[j];
        bq[qi][f][j + 4]  = (_Float16)a1[j];
        bq[qi][f][j + 8]  = (_Float16)a2[j];
        bq[qi][f][j + 12] = (_Float16)a3[j];
      }
    }
  }

  v8f acc[2][4];
#pragma unroll
  for (int qi = 0; qi < 2; ++qi)
    for (int t = 0; t < 4; ++t)
      for (int r = 0; r < 8; ++r) acc[qi][t][r] = 0.0f;
  float m_i[2] = {NEG, NEG};
  float l_i[2] = {0.0f, 0.0f};     // own-half partial sums; merged in epilogue

  const int nkb = (qbase >> 5) + 4;

  // ---- prologue: issue block-0 loads (8x global_load_b128, no waits) ----
  v4f rk[4], rv[4];
#pragma unroll
  for (int i = 0; i < 4; ++i) {
    const size_t g = (size_t)kk[i] * Hh * Dd + dd[i];
    rk[i] = *(const v4f*)(Kb + g);
    rv[i] = *(const v4f*)(Vb + g);
  }

  for (int ib = 0; ib < nkb; ++ib) {
    const int cur = ib & 1;
    const int kb  = ib * KB;

    // ---- stage current block into LDS (f32 -> f16) ----
#pragma unroll
    for (int i = 0; i < 4; ++i) {
      v4h hk;
#pragma unroll
      for (int j = 0; j < 4; ++j) hk[j] = (_Float16)rk[i][j];
      *(v4h*)(&sK[cur][kk[i] * KROW + dd[i]]) = hk;      // ds_store_b64
#pragma unroll
      for (int j = 0; j < 4; ++j)
        sV[cur][(dd[i] + j) * VROW + kk[i]] = (_Float16)rv[i][j];
    }

    // ---- issue next block's loads before the barrier (latency overlap) ----
    if (ib + 1 < nkb) {
      const int kn = kb + KB;
#pragma unroll
      for (int i = 0; i < 4; ++i) {
        const size_t g = (size_t)(kn + kk[i]) * Hh * Dd + dd[i];
        rk[i] = *(const v4f*)(Kb + g);
        rv[i] = *(const v4f*)(Vb + g);
      }
    }

    __syncthreads();   // single barrier/iter: double buffer makes it safe

    // both q-tiles active iff kb <= qt0 (kb, qt0 are 32-aligned); uniform.
    if (kb <= qt0) {
      const _Float16* cK = sK[cur];
      const _Float16* cV = sV[cur];

      // ---- K A-fragments, shared by both q-tiles ----
      v16h aK[2][2];                               // [ktile][dh]
#pragma unroll
      for (int kt = 0; kt < 2; ++kt)
#pragma unroll
        for (int dh = 0; dh < 2; ++dh)
          aK[kt][dh] = frag(cK, (kt * 16 + ll) * KROW + 32 * dh + 8 * hb,
                                (kt * 16 + ll) * KROW + 32 * dh + 16 + 8 * hb);

      // ---- S^T = K x Q^T : 8 WMMAs, 4 independent accumulator chains ----
      v8f s[2][2];                                 // [qi][ktile]
#pragma unroll
      for (int qi = 0; qi < 2; ++qi)
        for (int kt = 0; kt < 2; ++kt)
          for (int r = 0; r < 8; ++r) s[qi][kt][r] = 0.0f;
#pragma unroll
      for (int dh = 0; dh < 2; ++dh)
#pragma unroll
        for (int qi = 0; qi < 2; ++qi)
#pragma unroll
          for (int kt = 0; kt < 2; ++kt)
            s[qi][kt] = __builtin_amdgcn_wmma_f32_16x16x32_f16(
                false, aK[kt][dh], false, bq[qi][dh], (short)0, s[qi][kt],
                false, false);

      // ---- per-tile softmax; masking only on the diagonal block ----
      v16h bp[2];
      const bool diag = (kb == qt0);
#pragma unroll
      for (int qi = 0; qi < 2; ++qi) {
        const int q = qt0 + qi * 16 + ll;
        if (diag) {
#pragma unroll
          for (int r = 0; r < 8; ++r) {
            const int k0 = kb + r + 8 * hb;
            s[qi][0][r] = (k0      > q) ? NEG : s[qi][0][r];
            s[qi][1][r] = (k0 + 16 > q) ? NEG : s[qi][1][r];
          }
        }
        // row max MUST merge the lane pair (both halves feed one column)
        float vmax = NEG;
#pragma unroll
        for (int r = 0; r < 8; ++r)
          vmax = fmaxf(vmax, fmaxf(s[qi][0][r], s[qi][1][r]));
        vmax = fmaxf(vmax, __shfl_xor(vmax, 16, 32));
        const float m_new = fmaxf(m_i[qi], vmax);
        const float ms    = m_new * SCL;
        const float alpha = fexp2(fmaf(m_i[qi], SCL, -ms));

        float rsum = 0.0f;                      // own-half sum only (deferred)
#pragma unroll
        for (int r = 0; r < 8; ++r) {
          const float p0 = fexp2(fmaf(s[qi][0][r], SCL, -ms));
          const float p1 = fexp2(fmaf(s[qi][1][r], SCL, -ms));
          rsum += p0 + p1;
          bp[qi][r]     = (_Float16)p0;
          bp[qi][r + 8] = (_Float16)p1;
        }
        l_i[qi] = l_i[qi] * alpha + rsum;       // same alpha in both lanes
        m_i[qi] = m_new;
#pragma unroll
        for (int t = 0; t < 4; ++t)
          for (int r = 0; r < 8; ++r) acc[qi][t][r] *= alpha;
      }

      // ---- O^T += V^T x P^T : V fragments reused by both q-tiles ----
#pragma unroll
      for (int t = 0; t < 4; ++t) {
        const int d = t * 16 + ll;
        v16h av = frag(cV, d * VROW + 8 * hb, d * VROW + 16 + 8 * hb);
        acc[0][t] = __builtin_amdgcn_wmma_f32_16x16x32_f16(
            false, av, false, bp[0], (short)0, acc[0][t], false, false);
        acc[1][t] = __builtin_amdgcn_wmma_f32_16x16x32_f16(
            false, av, false, bp[1], (short)0, acc[1][t], false, false);
      }
    }
  }

  // ---- epilogue: merge partner half-sums once, then normalize ----
#pragma unroll
  for (int qi = 0; qi < 2; ++qi) {
    const float l_tot = l_i[qi] + __shfl_xor(l_i[qi], 16, 32);
    const float inv = 1.0f / l_tot;
    float* orow = Ob + (size_t)(qt0 + qi * 16 + ll) * Hh * Dd;
#pragma unroll
    for (int t = 0; t < 4; ++t) {
      v4f o0, o1;
#pragma unroll
      for (int r = 0; r < 4; ++r) {
        o0[r] = acc[qi][t][r] * inv;
        o1[r] = acc[qi][t][r + 4] * inv;
      }
      *(v4f*)(orow + t * 16 + 8 * hb)     = o0;
      *(v4f*)(orow + t * 16 + 8 * hb + 4) = o1;
    }
  }
}

extern "C" void kernel_launch(void* const* d_in, const int* in_sizes, int n_in,
                              void* d_out, int out_size, void* d_ws, size_t ws_size,
                              hipStream_t stream) {
  const float* q = (const float*)d_in[0];
  const float* k = (const float*)d_in[1];
  const float* v = (const float*)d_in[2];
  float* o = (float*)d_out;
  dim3 grid(Ls / BLOCK_M, Bb * Hh);   // 32 x 16 = 512 workgroups
  fa_fwd_causal<<<grid, dim3(128), 0, stream>>>(q, k, v, o);
}
